// MSDeformAttn_1717986918603
// MI455X (gfx1250) — matrix operand order repaired and checked by reading, not compiled, not run on hardware
//
#include <hip/hip_runtime.h>
#include <hip/hip_bf16.h>

typedef __attribute__((ext_vector_type(16))) __bf16 v16bf;
typedef __attribute__((ext_vector_type(8)))  float  v8f;

// types matching __builtin_amdgcn_global_load_async_to_lds_b128's prototype:
// (int4 addrspace(1)*, int4 addrspace(3)*, imm offset, imm cpol)
typedef int v4i_ __attribute__((vector_size(16)));
typedef __attribute__((address_space(1))) v4i_ gv4i_t;   // global
typedef __attribute__((address_space(3))) v4i_ lv4i_t;   // LDS

#define LQ      12240
#define LEN_IN  12240
#define NB      2
#define CDIM    256
#define KDIM    256          // K of every GEMM in this op
#define MHEADS  8
#define DHEAD   32
#define NLQ     (NB * LQ)    // 24480 rows, divisible by 16

union ABf  { v16bf v; uint4 u[2]; };

// ---------------------------------------------------------------------------
// One-time f32 -> bf16 pack of activations (8 elements / thread).
// ---------------------------------------------------------------------------
__global__ __launch_bounds__(256) void cvt_f32_to_bf16_kernel(
    const float* __restrict__ s, __bf16* __restrict__ d, int n8)
{
    int i = blockIdx.x * blockDim.x + threadIdx.x;
    if (i >= n8) return;
    const float4* sp = (const float4*)(s + (size_t)i * 8);
    float4 x0 = sp[0], x1 = sp[1];
    union { __bf16 b[8]; uint4 u; } o;
    o.b[0] = (__bf16)x0.x; o.b[1] = (__bf16)x0.y;
    o.b[2] = (__bf16)x0.z; o.b[3] = (__bf16)x0.w;
    o.b[4] = (__bf16)x1.x; o.b[5] = (__bf16)x1.y;
    o.b[6] = (__bf16)x1.z; o.b[7] = (__bf16)x1.w;
    *(uint4*)(d + (size_t)i * 8) = o.u;
}

// ---------------------------------------------------------------------------
// One-time weight repack: f32 row-major (K, Ntot) -> bf16 WMMA B-fragment
// panel order. Panel tn (16 columns) is 4096 contiguous bf16 (8 KB):
//   Wp[tn*4096 + kb*512 + ln*32 + kj] = W[(kb*32 + kj) * Ntot + tn*16 + ln]
// so the GEMM stages each panel into LDS with a *contiguous* copy and lane
// (g,ln) reads its fragment for K-block kb as one 32-byte run.
// ---------------------------------------------------------------------------
__global__ __launch_bounds__(256) void pack_w_bf16_kernel(
    const float* __restrict__ W, __bf16* __restrict__ Wp, int Ntot)
{
    int i = blockIdx.x * blockDim.x + threadIdx.x;
    if (i >= KDIM * Ntot) return;
    int tn  = i >> 12;           // panel
    int rem = i & 4095;
    int kb  = rem >> 9;
    int r2  = rem & 511;
    int ln  = r2 >> 5;
    int kj  = r2 & 31;
    Wp[i] = (__bf16)W[(size_t)(kb * 32 + kj) * Ntot + tn * 16 + ln];
}

// ---------------------------------------------------------------------------
// WMMA GEMM: Y[M,N](f32) = X[M,K](bf16) @ W(packed bf16 panels) + bias[N]
// Block = 256 threads = 8 waves = 128x16 macro-tile.
// The 8 KB weight panel is staged global->LDS via global_load_async_to_lds_b128
// (ASYNCcnt) when available, else via vector b128 copies. Fragments then come
// from LDS as 2 x ds_load_b128, A as 2 x global_load_b128, 8 WMMAs per tile.
// ---------------------------------------------------------------------------
__global__ __launch_bounds__(256) void wmma_gemm_bf16_kernel(
    const __bf16* __restrict__ X, const __bf16* __restrict__ Wp,
    const float* __restrict__ bias, float* __restrict__ Y,
    int Mtot, int Ntot)
{
    __shared__ __bf16 ldsB[8 * 512];            // one 8 KB panel

    const int col0 = blockIdx.y * 16;
    const int tid  = threadIdx.x;

    // ---- stage panel: contiguous 8192-byte copy, 32 bytes per thread ----
    {
        const __bf16* gsrc = Wp + (size_t)blockIdx.y * 4096 + (size_t)tid * 16;
        __bf16*       ldst = &ldsB[tid * 16];
#if __has_builtin(__builtin_amdgcn_global_load_async_to_lds_b128)
        // each call copies 16 bytes global->LDS at offset off; tracked by ASYNCcnt
        gv4i_t* gp = (gv4i_t*)gsrc;
        lv4i_t* lp = (lv4i_t*)ldst;
        __builtin_amdgcn_global_load_async_to_lds_b128(gp, lp, 0, 0);
        __builtin_amdgcn_global_load_async_to_lds_b128(gp, lp, 16, 0);
#if __has_builtin(__builtin_amdgcn_s_wait_asynccnt)
        __builtin_amdgcn_s_wait_asynccnt(0);
#else
        asm volatile("s_wait_asynccnt 0x0" ::: "memory");
#endif
#else
        const uint4* gs = (const uint4*)gsrc;
        uint4*       ls = (uint4*)ldst;
        ls[0] = gs[0];
        ls[1] = gs[1];
#endif
    }
    __syncthreads();

    const int lane = tid & 31;
    const int g    = lane >> 4;
    const int ln   = lane & 15;
    const int row0 = (blockIdx.x * 8 + (tid >> 5)) * 16;
    if (row0 >= Mtot) return;                   // wave-uniform tail guard

    const __bf16* xrow = X + (size_t)(row0 + ln) * KDIM;
    v8f acc = {};
    #pragma unroll
    for (int kb = 0; kb < 8; ++kb) {
        ABf a, b;
        const uint4* ap = (const uint4*)(xrow + kb * 32 + g * 16);
        a.u[0] = ap[0];                         // K = kb*32 + 16g     .. +7
        a.u[1] = ap[1];                         // K = kb*32 + 16g + 8 .. +15
        const uint4* bp = (const uint4*)&ldsB[kb * 512 + ln * 32 + g * 16];
        b.u[0] = bp[0];                         // K = kb*32 + 16g     .. +7
        b.u[1] = bp[1];                         // K = kb*32 + 16g + 8 .. +15
        acc = __builtin_amdgcn_wmma_f32_16x16x32_bf16(
            false, a.v, false, b.v, (short)0, acc, false, false);
    }

    float  bcol = bias[col0 + ln];
    float* yp   = Y + (size_t)(row0 + g * 8) * Ntot + col0 + ln;
    #pragma unroll
    for (int r = 0; r < 8; ++r)                 // C/D: VGPR r -> rows (r, r+8)
        yp[(size_t)r * Ntot] = acc[r] + bcol;
}

// ---------------------------------------------------------------------------
// Softmax over groups of 16 consecutive floats (L*P = 16 per head).
// ---------------------------------------------------------------------------
__global__ __launch_bounds__(256) void softmax16_kernel(float* __restrict__ a, int ngroups)
{
    int r = blockIdx.x * blockDim.x + threadIdx.x;
    if (r >= ngroups) return;
    float* p = a + (size_t)r * 16;
    float v[16], mx = -3.402823466e+38f;
    #pragma unroll
    for (int i = 0; i < 16; ++i) { v[i] = p[i]; mx = fmaxf(mx, v[i]); }
    float s = 0.f;
    #pragma unroll
    for (int i = 0; i < 16; ++i) { v[i] = __expf(v[i] - mx); s += v[i]; }
    float inv = 1.0f / s;
    #pragma unroll
    for (int i = 0; i < 16; ++i) p[i] = v[i] * inv;
}

// ---------------------------------------------------------------------------
// Deformable bilinear sampling. One wave per (n, q, m); lane = channel d.
// value (N, LEN_IN, M, D) f32: D contiguous -> each tap is a coalesced
// 128-byte wave read from the L2-resident value tensor. Output written
// directly in bf16 so the final WMMA GEMM consumes it with no extra pass.
// ---------------------------------------------------------------------------
__device__ __forceinline__ float tap(const float* __restrict__ vb,
                                     int st, int Wl, int Hl, int xx, int yy)
{
    if (xx < 0 || xx >= Wl || yy < 0 || yy >= Hl) return 0.f;
    return vb[(size_t)(st + yy * Wl + xx) * (MHEADS * DHEAD)];
}

__global__ __launch_bounds__(256) void msda_sample_kernel(
    const float* __restrict__ value,    // (N, LEN_IN, M, D) f32
    const float* __restrict__ refpts,   // (N, Lq, 4, 2) f32
    const float* __restrict__ offs,     // (N*Lq, M*L*P*2) f32
    const float* __restrict__ attn,     // (N*Lq, M*L*P) f32, softmaxed
    __bf16* __restrict__ out)           // (N*Lq, M*D) bf16
{
    const int Hs[4] = {96, 48, 24, 12};
    const int Ws[4] = {96, 48, 24, 12};
    const int ST[4] = {0, 9216, 11520, 12096};

    int wave = blockIdx.x * (blockDim.x >> 5) + (threadIdx.x >> 5);
    int lane = threadIdx.x & 31;
    int m  = wave & (MHEADS - 1);
    int nq = wave >> 3;                         // n*Lq + q
    if (nq >= NLQ) return;
    int n = nq / LQ;

    const float* ref = refpts + (size_t)nq * 8;
    const float* off = offs   + (size_t)nq * 256 + m * 32;
    const float* aw  = attn   + (size_t)nq * 128 + m * 16;
    const float* vb  = value  + (size_t)n * LEN_IN * CDIM + m * DHEAD + lane;

    float acc = 0.f;
    #pragma unroll
    for (int l = 0; l < 4; ++l) {
        float Wf = (float)Ws[l], Hf = (float)Hs[l];
        float rx = ref[l * 2 + 0], ry = ref[l * 2 + 1];
        #pragma unroll
        for (int p = 0; p < 4; ++p) {
            // (ref + off/norm)*dim - 0.5 == ref*dim + off - 0.5 (norm == dim)
            float x = rx * Wf + off[l * 8 + p * 2 + 0] - 0.5f;
            float y = ry * Hf + off[l * 8 + p * 2 + 1] - 0.5f;
            float x0f = floorf(x), y0f = floorf(y);
            int   x0  = (int)x0f,  y0  = (int)y0f;
            float wx1 = x - x0f, wx0 = 1.f - wx1;
            float wy1 = y - y0f, wy0 = 1.f - wy1;
            float a = aw[l * 4 + p];
            float v00 = tap(vb, ST[l], Ws[l], Hs[l], x0,     y0);
            float v10 = tap(vb, ST[l], Ws[l], Hs[l], x0 + 1, y0);
            float v01 = tap(vb, ST[l], Ws[l], Hs[l], x0,     y0 + 1);
            float v11 = tap(vb, ST[l], Ws[l], Hs[l], x0 + 1, y0 + 1);
            acc += a * (wx0 * wy0 * v00 + wx1 * wy0 * v10 +
                        wx0 * wy1 * v01 + wx1 * wy1 * v11);
        }
    }
    out[(size_t)nq * CDIM + m * DHEAD + lane] = (__bf16)acc;
}

// ---------------------------------------------------------------------------
extern "C" void kernel_launch(void* const* d_in, const int* in_sizes, int n_in,
                              void* d_out, int out_size, void* d_ws, size_t ws_size,
                              hipStream_t stream)
{
    const float* query  = (const float*)d_in[0];
    const float* refpts = (const float*)d_in[1];
    const float* inflat = (const float*)d_in[2];
    // d_in[3], d_in[4]: spatial shapes / level starts (compile-time constants)
    const float* Wv   = (const float*)d_in[5];
    const float* bv   = (const float*)d_in[6];
    const float* Wo   = (const float*)d_in[7];
    const float* bo   = (const float*)d_in[8];
    const float* Wa   = (const float*)d_in[9];
    const float* ba   = (const float*)d_in[10];
    const float* Wout = (const float*)d_in[11];
    const float* bout = (const float*)d_in[12];
    float* out = (float*)d_out;

    const size_t SZ_ACT = (size_t)NLQ * CDIM;       // 6,266,880 elements

    // ---- workspace carve-up (bytes) ----
    char* p = (char*)d_ws;
    float*  value = (float*)p;            p += SZ_ACT * 4;           // f32
    float*  offsb = (float*)p;            p += SZ_ACT * 4;           // f32
    float*  attnb = (float*)p;            p += (size_t)NLQ * 128 * 4;// f32
    __bf16* tmpb  = (__bf16*)p;           p += SZ_ACT * 2;           // bf16
    __bf16* xinb  = (__bf16*)p;           p += SZ_ACT * 2;           // bf16 input_flatten
    __bf16* qb    = (__bf16*)p;           p += SZ_ACT * 2;           // bf16 query
    __bf16* Wvb   = (__bf16*)p;           p += (size_t)256 * 256 * 2;// packed panels
    __bf16* Wob   = (__bf16*)p;           p += (size_t)256 * 256 * 2;
    __bf16* Wab   = (__bf16*)p;           p += (size_t)256 * 128 * 2;
    __bf16* Woutb = (__bf16*)p;           p += (size_t)256 * 256 * 2;

    // ---- one-time bf16 packing of GEMM operands ----
    {
        int n8 = (int)(SZ_ACT / 8);
        cvt_f32_to_bf16_kernel<<<(n8 + 255) / 256, 256, 0, stream>>>(inflat, xinb, n8);
        cvt_f32_to_bf16_kernel<<<(n8 + 255) / 256, 256, 0, stream>>>(query,  qb,   n8);
        pack_w_bf16_kernel<<<(256 * 256) / 256, 256, 0, stream>>>(Wv,   Wvb,   CDIM);
        pack_w_bf16_kernel<<<(256 * 256) / 256, 256, 0, stream>>>(Wo,   Wob,   CDIM);
        pack_w_bf16_kernel<<<(256 * 128) / 256, 256, 0, stream>>>(Wa,   Wab,   128);
        pack_w_bf16_kernel<<<(256 * 256) / 256, 256, 0, stream>>>(Wout, Woutb, CDIM);
    }

    const int tilesM = NLQ / 16;                    // 1530
    const int gx     = (tilesM + 7) / 8;            // 192 row strips

    // 1) value = input_flatten @ Wv + bv
    wmma_gemm_bf16_kernel<<<dim3(gx, CDIM / 16), 256, 0, stream>>>(
        xinb, Wvb, bv, value, NLQ, CDIM);
    // 2) offsets = query @ Wo + bo
    wmma_gemm_bf16_kernel<<<dim3(gx, CDIM / 16), 256, 0, stream>>>(
        qb, Wob, bo, offsb, NLQ, CDIM);
    // 3) attn = softmax(query @ Wa + ba) over groups of 16
    wmma_gemm_bf16_kernel<<<dim3(gx, 128 / 16), 256, 0, stream>>>(
        qb, Wab, ba, attnb, NLQ, 128);
    {
        int ngroups = NLQ * MHEADS;
        softmax16_kernel<<<(ngroups + 255) / 256, 256, 0, stream>>>(attnb, ngroups);
    }
    // 4) deformable sampling -> tmp (bf16)
    {
        int nwaves = NLQ * MHEADS;
        msda_sample_kernel<<<(nwaves + 7) / 8, 256, 0, stream>>>(
            value, refpts, offsb, attnb, tmpb);
    }
    // 5) out = tmp @ Wout + bout
    wmma_gemm_bf16_kernel<<<dim3(gx, CDIM / 16), 256, 0, stream>>>(
        tmpb, Woutb, bout, out, NLQ, CDIM);
}
